// VisualEncoderLayer_43422119362975
// MI455X (gfx1250) — compile-verified, matrix operand order checked by reading, not run on hardware
//
#include <hip/hip_runtime.h>
#include <hip/hip_bf16.h>
#include <math.h>

// ---- problem constants (fixed by the reference) ----
#define D     256
#define NH    8
#define HD    32
#define NL    4
#define NP    4
#define DFF   1024
#define BSZ   2
#define LT    20
#define LQv   20197
#define MROWS (BSZ * LQv)      // 40394 token rows

typedef unsigned short ushort_t;

// ---- WMMA / TDM types ----
typedef __attribute__((ext_vector_type(16))) __bf16 v16bf;
typedef __attribute__((ext_vector_type(8)))  float  v8f;
typedef __attribute__((ext_vector_type(4)))  unsigned int v4u;
typedef __attribute__((ext_vector_type(8)))  int v8i;
typedef __attribute__((ext_vector_type(4)))  int v4i;

union FragAB { v16bf v; unsigned int w[8]; ushort_t u[16]; };

__device__ __forceinline__ ushort_t f2bf(float f) {
  unsigned int x = __float_as_uint(f);
  x = x + 0x7FFFu + ((x >> 16) & 1u);     // round-to-nearest-even
  return (ushort_t)(x >> 16);
}

// =====================================================================
// Tiled GEMM: C[M,N] = A[M,K](bf16) * W[N,K](bf16)^T + bias[N]
// block = 256 threads = 8 waves; tile 128x128; BK=32.
// Wave grid 4x2: each wave computes 32x64 = 2x4 WMMA fragments.
// A tile staged by plain b128 LDS stores; W tile loaded by the
// Tensor Data Mover (tensor_load_to_lds) with LDS row padding done
// by the TDM pad feature (16 DW interval + 4 DW pad -> 80B rows).
// =====================================================================
#define BM  128
#define BN  128
#define BK  32
#define LDT 40   // LDS row stride in ushorts -> 80B (64B data + 16B pad)

__global__ __launch_bounds__(256) void gemm_bias_wmma(
    const ushort_t* __restrict__ A, const ushort_t* __restrict__ W,
    const float* __restrict__ bias, float* __restrict__ C,
    ushort_t* __restrict__ Cb, int M, int N, int K, int relu)
{
  __shared__ __align__(16) ushort_t sA[BM][LDT];
  __shared__ __align__(16) ushort_t sW[BN][LDT];

  const int tid  = threadIdx.x;
  const int lane = tid & 31;
  const int wave = tid >> 5;
  const int wm   = (wave & 3) << 5;   // 0,32,64,96
  const int wn   = (wave >> 2) << 6;  // 0,64
  const int bM   = blockIdx.y * BM;
  const int bN   = blockIdx.x * BN;
  const int l15  = lane & 15;
  const int kb   = (lane & 16) ? 8 : 0;   // half-wave K base per ISA A/B layout

  v8f acc[2][4] = {};

  const int seg8 = (tid & 3) << 3;   // 0,8,16,24 (ushort column, 16B chunks)
  const int r0   = tid >> 2;         // 0..63

  const unsigned ldsW = (unsigned)(unsigned long long)(const void*)&sW[0][0];

  for (int k0 = 0; k0 < K; k0 += BK) {
    // ---- stage A tile: 128x32 bf16, pure 16B copies ----
#pragma unroll
    for (int i = 0; i < 2; ++i) {
      int r  = r0 + (i << 6);
      int gr = bM + r;
      uint4 v = make_uint4(0u, 0u, 0u, 0u);
      if (gr < M) v = *(const uint4*)(A + (size_t)gr * K + k0 + seg8);
      *(uint4*)&sA[r][seg8] = v;
    }
    // ---- W tile via Tensor Data Mover (wave 0 issues; EXEC ignored) ----
    if (wave == 0) {
      unsigned long long ga =
          (unsigned long long)(const void*)(W + (size_t)bN * K + k0);
      v4u g0;
      g0.x = 1u;                                   // count=1 (valid), user D#
      g0.y = ldsW;                                 // lds_addr (bytes)
      g0.z = (unsigned)(ga & 0xffffffffu);         // global_addr[31:0]
      g0.w = (unsigned)((ga >> 32) & 0x1ffffffu)   // global_addr[56:32]
           | (2u << 30);                           // type=2 ("image")
      unsigned w0 = (1u << 16)   // data_size=1 -> 2 bytes
                  | (1u << 20)   // pad_enable
                  | (3u << 22)   // pad_interval: 16 DWORDs (one 64B row)
                  | (3u << 25);  // pad_amount: 4 DWORDs (16B)
      unsigned w1 = ((unsigned)K & 0xffffu) << 16;           // tensor_dim0 lo16
      unsigned w2 = ((unsigned)K >> 16) | (((unsigned)N & 0xffffu) << 16);
      unsigned w3 = ((unsigned)N >> 16) | (32u << 16);       // tile_dim0 = 32
      unsigned w4 = (unsigned)BN;                            // tile_dim1 = 128
      unsigned w5 = (unsigned)K;                             // dim0_stride lo32
      v8i g1 = { (int)w0, (int)w1, (int)w2, (int)w3,
                 (int)w4, (int)w5, 0, 0 };
      v4i z4 = { 0, 0, 0, 0 };
      v8i z8 = { 0, 0, 0, 0, 0, 0, 0, 0 };
      __builtin_amdgcn_tensor_load_to_lds(g0, g1, z4, z4, z8, 0);
      __builtin_amdgcn_s_wait_tensorcnt(0);
    }
    __syncthreads();

    // ---- fragment loads: two 16B LDS reads per operand ----
    FragAB a[2], b[4];
#pragma unroll
    for (int t = 0; t < 2; ++t) {
      int row = wm + (t << 4) + l15;
      uint4 L = *(const uint4*)&sA[row][kb];
      uint4 H = *(const uint4*)&sA[row][kb + 16];
      a[t].w[0] = L.x; a[t].w[1] = L.y; a[t].w[2] = L.z; a[t].w[3] = L.w;
      a[t].w[4] = H.x; a[t].w[5] = H.y; a[t].w[6] = H.z; a[t].w[7] = H.w;
    }
#pragma unroll
    for (int t = 0; t < 4; ++t) {
      int col = wn + (t << 4) + l15;
      uint4 L = *(const uint4*)&sW[col][kb];
      uint4 H = *(const uint4*)&sW[col][kb + 16];
      b[t].w[0] = L.x; b[t].w[1] = L.y; b[t].w[2] = L.z; b[t].w[3] = L.w;
      b[t].w[4] = H.x; b[t].w[5] = H.y; b[t].w[6] = H.z; b[t].w[7] = H.w;
    }
#pragma unroll
    for (int tm = 0; tm < 2; ++tm)
#pragma unroll
      for (int tn = 0; tn < 4; ++tn)
        acc[tm][tn] = __builtin_amdgcn_wmma_f32_16x16x32_bf16(
            false, a[tm].v, false, b[tn].v, (short)0, acc[tm][tn], false, false);
    __syncthreads();
  }

  // ---- epilogue: lane = column, vgpr r = row r (lanes 0-15) / 8+r ----
#pragma unroll
  for (int tm = 0; tm < 2; ++tm)
#pragma unroll
    for (int tn = 0; tn < 4; ++tn) {
      int col     = bN + wn + (tn << 4) + l15;
      int rowBase = bM + wm + (tm << 4) + ((lane & 16) ? 8 : 0);
      float bb = (col < N) ? bias[col] : 0.f;
#pragma unroll
      for (int r = 0; r < 8; ++r) {
        int row = rowBase + r;
        if (row < M && col < N) {
          float vv = acc[tm][tn][r] + bb;
          if (relu) vv = fmaxf(vv, 0.f);
          if (C)  C[(size_t)row * N + col] = vv;
          if (Cb) Cb[(size_t)row * N + col] = f2bf(vv);
        }
      }
    }
}

// =====================================================================
// conversion helpers
// =====================================================================
__global__ void cvt_bf16(const float* __restrict__ a, ushort_t* __restrict__ o, int n) {
  int i = blockIdx.x * blockDim.x + threadIdx.x;
  if (i < n) o[i] = f2bf(a[i]);
}
__global__ void addcvt_bf16(const float* __restrict__ a, const float* __restrict__ b,
                            ushort_t* __restrict__ o, int n) {
  int i = blockIdx.x * blockDim.x + threadIdx.x;
  if (i < n) o[i] = f2bf(a[i] + b[i]);
}
__global__ void vcopy(const float* __restrict__ a, float* __restrict__ o, int n) {
  int i = blockIdx.x * blockDim.x + threadIdx.x;
  if (i < n) o[i] = a[i];
}

// softmax over the NL*NP=16 sampling weights per (row, head), in place
__global__ void softmax16(float* __restrict__ aw, int total) {
  int i = blockIdx.x * blockDim.x + threadIdx.x;
  if (i >= total) return;
  float* p = aw + (size_t)i * 16;
  float m = -1e30f;
#pragma unroll
  for (int j = 0; j < 16; ++j) m = fmaxf(m, p[j]);
  float e[16], s = 0.f;
#pragma unroll
  for (int j = 0; j < 16; ++j) { e[j] = __expf(p[j] - m); s += e[j]; }
  float inv = 1.f / s;
#pragma unroll
  for (int j = 0; j < 16; ++j) p[j] = e[j] * inv;
}

// =====================================================================
// MSDeform bilinear sampling: one wave per query row, lanes span HD=32.
// value fits in L2 (41 MB < 192 MB) so corner gathers are 128B L2 hits.
// Emits bf16 directly (only consumer is the output-proj GEMM).
// =====================================================================
__global__ __launch_bounds__(256) void msdeform(
    const float* __restrict__ value,   // [MROWS, D]
    const float* __restrict__ offs,    // [MROWS, 256]
    const float* __restrict__ aw,      // [MROWS, 128] softmaxed
    const float* __restrict__ ref,     // [B, LQ, NL, 2]
    ushort_t* __restrict__ ms)         // [MROWS, D] bf16
{
  const int Hs[4] = {100, 50, 25, 13};
  const int Ws[4] = {152, 76, 38, 19};
  const int St[4] = {0, 15200, 19000, 19950};

  int r = blockIdx.x * 8 + (threadIdx.x >> 5);
  if (r >= MROWS) return;
  int lane = threadIdx.x & 31;          // d
  int b = r / LQv;
  size_t vbase = (size_t)b * LQv * D;
  size_t obase = (size_t)r * D;
  size_t abase = (size_t)r * (NH * NL * NP);
  size_t rbase = (size_t)r * (NL * 2);

  for (int h = 0; h < NH; ++h) {
    float acc = 0.f;
#pragma unroll
    for (int l = 0; l < NL; ++l) {
      float Wf = (float)Ws[l], Hf = (float)Hs[l];
      float rx = ref[rbase + l * 2 + 0];
      float ry = ref[rbase + l * 2 + 1];
#pragma unroll
      for (int p = 0; p < NP; ++p) {
        int oc = (((h * NL + l) * NP) + p) * 2;
        float lx = rx + offs[obase + oc + 0] / Wf;
        float ly = ry + offs[obase + oc + 1] / Hf;
        float x = lx * Wf - 0.5f;
        float y = ly * Hf - 0.5f;
        float xf = floorf(x), yf = floorf(y);
        float fx = x - xf, fy = y - yf;
        int x0 = (int)xf, y0 = (int)yf;
        float wgt = aw[abase + (h * NL + l) * NP + p];
#pragma unroll
        for (int c = 0; c < 4; ++c) {
          int dx = c & 1, dy = c >> 1;
          float cw = (dx ? fx : 1.f - fx) * (dy ? fy : 1.f - fy);
          int xi = x0 + dx, yi = y0 + dy;
          if (xi >= 0 && xi < Ws[l] && yi >= 0 && yi < Hs[l]) {
            int idx = St[l] + yi * Ws[l] + xi;
            acc += wgt * cw * value[vbase + (size_t)idx * D + h * HD + lane];
          }
        }
      }
    }
    ms[obase + h * HD + lane] = f2bf(acc);
  }
}

// =====================================================================
// Residual add + LayerNorm: one wave per row, shfl_xor tree reductions.
// Writes fp32 y; optionally bf16(y [+ pos]) for the next GEMM's A input.
// =====================================================================
__global__ __launch_bounds__(256) void add_ln(
    const float* __restrict__ a, const float* __restrict__ b,
    const float* __restrict__ g, const float* __restrict__ be,
    float* __restrict__ out, const float* __restrict__ pos,
    ushort_t* __restrict__ out_bf, int Mr)
{
  int row = blockIdx.x * 8 + (threadIdx.x >> 5);
  if (row >= Mr) return;
  int lane = threadIdx.x & 31;
  size_t base = (size_t)row * D;
  float x[8], s = 0.f;
#pragma unroll
  for (int i = 0; i < 8; ++i) {
    int c = (i << 5) + lane;
    x[i] = a[base + c] + b[base + c];
    s += x[i];
  }
#pragma unroll
  for (int o = 16; o > 0; o >>= 1) s += __shfl_xor(s, o, 32);
  float mean = s * (1.f / D);
  float vs = 0.f;
#pragma unroll
  for (int i = 0; i < 8; ++i) { float d0 = x[i] - mean; vs += d0 * d0; }
#pragma unroll
  for (int o = 16; o > 0; o >>= 1) vs += __shfl_xor(vs, o, 32);
  float inv = rsqrtf(vs * (1.f / D) + 1e-5f);
#pragma unroll
  for (int i = 0; i < 8; ++i) {
    int c = (i << 5) + lane;
    float y = (x[i] - mean) * inv * g[c] + be[c];
    out[base + c] = y;
    if (out_bf) {
      float t = y;
      if (pos) t += pos[base + c];
      out_bf[base + c] = f2bf(t);
    }
  }
}

// =====================================================================
// Cross attention vs 20 text tokens: block = 1 query, wave = 1 head.
// Emits bf16 ctx (only consumer is the mha-out GEMM).
// =====================================================================
__global__ __launch_bounds__(256) void cross_attn(
    const float* __restrict__ qh,   // [MROWS, D]
    const float* __restrict__ kh,   // [B*LT, D]
    const float* __restrict__ vh,   // [B*LT, D]
    ushort_t* __restrict__ ctx)     // [MROWS, D] bf16
{
  int r = blockIdx.x;
  if (r >= MROWS) return;
  int h = threadIdx.x >> 5;
  int d = threadIdx.x & 31;
  int b = r / LQv;
  float q = qh[(size_t)r * D + h * HD + d];
  float sc[LT];
  float m = -1e30f;
#pragma unroll
  for (int k = 0; k < LT; ++k) {
    float s = q * kh[(size_t)(b * LT + k) * D + h * HD + d];
#pragma unroll
    for (int o = 16; o > 0; o >>= 1) s += __shfl_xor(s, o, 32);
    s *= 0.1767766953f;     // 1/sqrt(HD)
    sc[k] = s;
    m = fmaxf(m, s);
  }
  float sum = 0.f;
#pragma unroll
  for (int k = 0; k < LT; ++k) { sc[k] = __expf(sc[k] - m); sum += sc[k]; }
  float inv = 1.f / sum;
  float c = 0.f;
#pragma unroll
  for (int k = 0; k < LT; ++k)
    c += sc[k] * vh[(size_t)(b * LT + k) * D + h * HD + d];
  ctx[(size_t)r * D + h * HD + d] = f2bf(c * inv);
}

// =====================================================================
// launch
// =====================================================================
extern "C" void kernel_launch(void* const* d_in, const int* in_sizes, int n_in,
                              void* d_out, int out_size, void* d_ws, size_t ws_size,
                              hipStream_t stream) {
  (void)in_sizes; (void)n_in; (void)out_size; (void)ws_size;
  const float* src  = (const float*)d_in[0];
  const float* pos  = (const float*)d_in[1];
  const float* ref  = (const float*)d_in[2];
  const float* text = (const float*)d_in[5];
  const float* so_w = (const float*)d_in[7];
  const float* so_b = (const float*)d_in[8];
  const float* aw_w = (const float*)d_in[9];
  const float* aw_b = (const float*)d_in[10];
  const float* vp_w = (const float*)d_in[11];
  const float* vp_b = (const float*)d_in[12];
  const float* op_w = (const float*)d_in[13];
  const float* op_b = (const float*)d_in[14];
  const float* ln1g = (const float*)d_in[15];
  const float* ln1b = (const float*)d_in[16];
  const float* ipw  = (const float*)d_in[17];
  const float* ipb  = (const float*)d_in[18];
  const float* mow  = (const float*)d_in[19];
  const float* mob  = (const float*)d_in[20];
  const float* ln3g = (const float*)d_in[21];
  const float* ln3b = (const float*)d_in[22];
  const float* l1w  = (const float*)d_in[23];
  const float* l1b  = (const float*)d_in[24];
  const float* l2w  = (const float*)d_in[25];
  const float* l2b  = (const float*)d_in[26];
  const float* ln2g = (const float*)d_in[27];
  const float* ln2b = (const float*)d_in[28];
  float* out = (float*)d_out;

  const size_t R  = (size_t)MROWS * D;       // 10,340,864
  const int    TX = BSZ * LT * D;            // 10,240

  // ---- workspace layout ----
  float* ws = (float*)d_ws;
  float* P   = ws;                           // value -> ms2 -> qh -> mo -> x2
  float* Q   = P + R;                        // offs -> x -> f2
  float* AW  = Q + R;                        // [MROWS,128]
  float* KH  = AW + (size_t)MROWS * 128;
  float* VH  = KH + TX;
  ushort_t* src_bf  = (ushort_t*)(VH + TX);  // R
  ushort_t* qbf     = src_bf + R;            // q -> xq -> x2bf
  ushort_t* msbf    = qbf + R;               // ms -> ctx
  ushort_t* F1bf    = msbf + R;              // [MROWS, DFF] = 4R
  ushort_t* text_bf = F1bf + 4 * R;
  ushort_t* wb      = text_bf + TX;          // packed bf16 weights
  ushort_t* wb_vp = wb;                      // 64K each unless noted
  ushort_t* wb_so = wb_vp + D * D;
  ushort_t* wb_aw = wb_so + D * D;           // 32K
  ushort_t* wb_op = wb_aw + 128 * D;
  ushort_t* wb_ip = wb_op + D * D;           // 3*64K (q,k,v)
  ushort_t* wb_mo = wb_ip + 3 * D * D;
  ushort_t* wb_l1 = wb_mo + D * D;           // 256K
  ushort_t* wb_l2 = wb_l1 + DFF * D;         // 256K

  dim3 blk(256);
  const int rowBlocks = (MROWS + 7) / 8;
  auto grid1 = [](int n) { return dim3((n + 255) / 256); };
  auto gemmGrid = [](int M, int N) { return dim3((N + BN - 1) / BN, (M + BM - 1) / BM); };

  // ---- one-time bf16 conversions ----
  cvt_bf16<<<grid1(D * D), blk, 0, stream>>>(vp_w, wb_vp, D * D);
  cvt_bf16<<<grid1(D * D), blk, 0, stream>>>(so_w, wb_so, D * D);
  cvt_bf16<<<grid1(128 * D), blk, 0, stream>>>(aw_w, wb_aw, 128 * D);
  cvt_bf16<<<grid1(D * D), blk, 0, stream>>>(op_w, wb_op, D * D);
  cvt_bf16<<<grid1(3 * D * D), blk, 0, stream>>>(ipw, wb_ip, 3 * D * D);
  cvt_bf16<<<grid1(D * D), blk, 0, stream>>>(mow, wb_mo, D * D);
  cvt_bf16<<<grid1(DFF * D), blk, 0, stream>>>(l1w, wb_l1, DFF * D);
  cvt_bf16<<<grid1(DFF * D), blk, 0, stream>>>(l2w, wb_l2, DFF * D);
  cvt_bf16<<<grid1((int)R), blk, 0, stream>>>(src, src_bf, (int)R);
  cvt_bf16<<<grid1(TX), blk, 0, stream>>>(text, text_bf, TX);
  // q = src + pos (bf16 only; q feeds only GEMMs)
  addcvt_bf16<<<grid1((int)R), blk, 0, stream>>>(src, pos, qbf, (int)R);

  // ---- deformable branch ----
  gemm_bias_wmma<<<gemmGrid(MROWS, D), blk, 0, stream>>>(src_bf, wb_vp, vp_b, P, nullptr, MROWS, D, D, 0);
  gemm_bias_wmma<<<gemmGrid(MROWS, D), blk, 0, stream>>>(qbf, wb_so, so_b, Q, nullptr, MROWS, D, D, 0);
  gemm_bias_wmma<<<gemmGrid(MROWS, 128), blk, 0, stream>>>(qbf, wb_aw, aw_b, AW, nullptr, MROWS, 128, D, 0);
  softmax16<<<grid1(MROWS * NH), blk, 0, stream>>>(AW, MROWS * NH);
  msdeform<<<dim3(rowBlocks), blk, 0, stream>>>(P, Q, AW, ref, msbf);
  gemm_bias_wmma<<<gemmGrid(MROWS, D), blk, 0, stream>>>(msbf, wb_op, op_b, P, nullptr, MROWS, D, D, 0);
  // x = LN(src + ms2) -> Q ; xq_bf = bf16(x + pos) -> qbf
  add_ln<<<dim3(rowBlocks), blk, 0, stream>>>(src, P, ln1g, ln1b, Q, pos, qbf, MROWS);

  // ---- text cross-attention ----
  gemm_bias_wmma<<<gemmGrid(BSZ * LT, D), blk, 0, stream>>>(text_bf, wb_ip + (size_t)D * D, ipb + D, KH, nullptr, BSZ * LT, D, D, 0);
  gemm_bias_wmma<<<gemmGrid(BSZ * LT, D), blk, 0, stream>>>(text_bf, wb_ip + (size_t)2 * D * D, ipb + 2 * D, VH, nullptr, BSZ * LT, D, D, 0);
  gemm_bias_wmma<<<gemmGrid(MROWS, D), blk, 0, stream>>>(qbf, wb_ip, ipb, P, nullptr, MROWS, D, D, 0);
  cross_attn<<<dim3(MROWS), blk, 0, stream>>>(P, KH, VH, msbf);
  gemm_bias_wmma<<<gemmGrid(MROWS, D), blk, 0, stream>>>(msbf, wb_mo, mob, P, nullptr, MROWS, D, D, 0);
  // x2 = LN(x + mo) -> P ; x2_bf -> qbf
  add_ln<<<dim3(rowBlocks), blk, 0, stream>>>(Q, P, ln3g, ln3b, P, nullptr, qbf, MROWS);

  // ---- FFN ----
  gemm_bias_wmma<<<gemmGrid(MROWS, DFF), blk, 0, stream>>>(qbf, wb_l1, l1b, nullptr, F1bf, MROWS, DFF, D, 1);
  gemm_bias_wmma<<<gemmGrid(MROWS, D), blk, 0, stream>>>(F1bf, wb_l2, l2b, Q, nullptr, MROWS, D, DFF, 0);
  // out = LN(x2 + f2) -> d_out
  add_ln<<<dim3(rowBlocks), blk, 0, stream>>>(P, Q, ln2g, ln2b, out, nullptr, nullptr, MROWS);

  // second tuple output: text_memory passthrough
  vcopy<<<grid1(TX), blk, 0, stream>>>(text, out + R, TX);
}
